// WindGuidedCrossAttentionLayer_47184510714606
// MI455X (gfx1250) — matrix operand order repaired and checked by reading, not compiled, not run on hardware
//
#include <hip/hip_runtime.h>

// ---------------------------------------------------------------------------
// WindGuidedCrossAttentionLayer for MI455X (gfx1250), bf16 WMMA pipeline.
//   LN -> {Q,K,V} GEMMs -> flash attention (fused beta*wind_bias) -> Wo GEMM
//   (+residual) -> LN -> W1 GEMM (+GELU) -> W2 GEMM (+residual) -> out.
// All matmuls: v_wmma_f32_16x16x32_bf16 (f32 accum). Weights pre-transposed to
// [N][K] so every hot-loop LDS fill is a contiguous b128 copy (async-to-LDS
// when the toolchain exposes the gfx1250 builtin).
// ---------------------------------------------------------------------------

typedef __attribute__((ext_vector_type(16))) __bf16 v16bf;
typedef __attribute__((ext_vector_type(8)))  __bf16 v8bf;
typedef __attribute__((ext_vector_type(8)))  float  v8f;
typedef __attribute__((__vector_size__(4 * sizeof(int)))) int v4i;

#define DIM     768
#define HEADS   12
#define HD      64
#define HIDDEN  3072
#define BATCH   4
#define NQ      1024
#define NC      1024
#define MTOT    (BATCH * NQ)      // 4096 token rows

static __device__ __forceinline__ v8f wmma_bf16(v16bf a, v16bf b, v8f c) {
    return __builtin_amdgcn_wmma_f32_16x16x32_bf16(
        /*neg_a=*/false, a, /*neg_b=*/false, b,
        /*c_mod=*/(short)0, c, /*reuse_a=*/false, /*reuse_b=*/false);
}

// --- async global -> LDS b128 copy (gfx1250 path), with portable fallback ---
#if defined(__gfx1250__) && __has_builtin(__builtin_amdgcn_global_load_async_to_lds_b128)
#define HAVE_ASYNC_LDS 1
#endif

static __device__ __forceinline__ void g2l_b128(__bf16* lds, const __bf16* g) {
#if defined(HAVE_ASYNC_LDS)
    __builtin_amdgcn_global_load_async_to_lds_b128(
        (__attribute__((address_space(1))) v4i*)(unsigned long long)(uintptr_t)g,
        (__attribute__((address_space(3))) v4i*)(unsigned)(uintptr_t)lds,
        /*offset=*/0, /*cpol=*/0);
#else
    *(v8bf*)lds = *(const v8bf*)g;
#endif
}

static __device__ __forceinline__ void wait_lds_copies() {
#if defined(HAVE_ASYNC_LDS) && __has_builtin(__builtin_amdgcn_s_wait_asynccnt)
    __builtin_amdgcn_s_wait_asynccnt(0);
#endif
}

// A fragment (16x32 bf16, M x K): lane L holds row m=row0+(L&15);
// K = kb..kb+15, kb = (L<16?0:16).
// => two contiguous 16B LDS reads per lane.
static __device__ __forceinline__ v16bf frag_a(const __bf16* lds, int row0,
                                               int rs, int k0, int lane) {
    int m  = row0 + (lane & 15);
    int kb = k0 + ((lane & 16) ? 16 : 0);
    v8bf lo = *(const v8bf*)(lds + m * rs + kb);
    v8bf hi = *(const v8bf*)(lds + m * rs + kb + 8);
    v16bf r;
#pragma unroll
    for (int i = 0; i < 8; ++i) { r[i] = lo[i]; r[i + 8] = hi[i]; }
    return r;
}

// B fragment (32x16 bf16, K x N) from a K-major LDS tile tile[n][k]:
// lane L holds col n=n0+(L&15); K = kb..kb+15, kb=(L<16?0:16).
// => one contiguous 32B LDS read per lane.
static __device__ __forceinline__ v16bf frag_b(const __bf16* ldsT, int n0,
                                               int rs, int k0, int lane) {
    int n  = n0 + (lane & 15);
    int kb = k0 + ((lane & 16) ? 16 : 0);
    return *(const v16bf*)(ldsT + n * rs + kb);
}

// ---------------------------------------------------------------------------
// Tiled transpose + f32->bf16: out[N][K] = bf16(in[K][N]). Block (32,8).
// ---------------------------------------------------------------------------
__global__ void __launch_bounds__(256) k_transpose_bf16(const float* __restrict__ in,
                                                        __bf16* __restrict__ out,
                                                        int K, int N) {
    __shared__ float t[32][33];
    const int tx = threadIdx.x, ty = threadIdx.y;
    const int n0 = blockIdx.x * 32, k0 = blockIdx.y * 32;
#pragma unroll
    for (int j = 0; j < 4; ++j)
        t[ty + 8 * j][tx] = in[(size_t)(k0 + ty + 8 * j) * N + n0 + tx];
    __syncthreads();
#pragma unroll
    for (int j = 0; j < 4; ++j)
        out[(size_t)(n0 + ty + 8 * j) * K + k0 + tx] = (__bf16)t[tx][ty + 8 * j];
}

// ---------------------------------------------------------------------------
// LayerNorm over last dim (768), output bf16. One block per row.
// ---------------------------------------------------------------------------
__global__ void __launch_bounds__(256) k_layernorm_bf16(const float* __restrict__ x,
                                                        const float* __restrict__ w,
                                                        const float* __restrict__ bvec,
                                                        __bf16* __restrict__ out) {
    const int tid = threadIdx.x;
    const float* xr = x + (size_t)blockIdx.x * DIM;
    float v0 = xr[tid], v1 = xr[tid + 256], v2 = xr[tid + 512];
    __shared__ float rs[256], rq[256];
    rs[tid] = v0 + v1 + v2;
    rq[tid] = v0 * v0 + v1 * v1 + v2 * v2;
    __syncthreads();
#pragma unroll
    for (int o = 128; o > 0; o >>= 1) {
        if (tid < o) { rs[tid] += rs[tid + o]; rq[tid] += rq[tid + o]; }
        __syncthreads();
    }
    float mu   = rs[0] * (1.0f / DIM);
    float var  = rq[0] * (1.0f / DIM) - mu * mu;
    float rsig = rsqrtf(var + 1e-5f);
    __bf16* orow = out + (size_t)blockIdx.x * DIM;
    orow[tid]       = (__bf16)((v0 - mu) * rsig * w[tid]       + bvec[tid]);
    orow[tid + 256] = (__bf16)((v1 - mu) * rsig * w[tid + 256] + bvec[tid + 256]);
    orow[tid + 512] = (__bf16)((v2 - mu) * rsig * w[tid + 512] + bvec[tid + 512]);
}

// ---------------------------------------------------------------------------
// GEMM: C[M,N] = A[M,K] (bf16 row-major) @ W (given as WT[N][K] bf16) + bias
// Block tile 64x128, 8 waves in 2x4 grid, each wave 32x32 (4 WMMA / k-step).
// MODE 0: out_bf = bf16(C+bias)
// MODE 1: out_f  = aux + (C+bias)                 (residual, f32 out)
// MODE 2: out_bf = bf16(gelu_exact(C+bias))
// MODE 3: out_bf = bf16(C+bias) scattered to V-transposed [b,h,d,tok] layout
// ---------------------------------------------------------------------------
template <int MODE>
__global__ void __launch_bounds__(256) k_gemm_bf16(const __bf16* __restrict__ A,
                                                   const __bf16* __restrict__ WT,
                                                   const float* __restrict__ bias,
                                                   const float* __restrict__ aux,
                                                   float* __restrict__ outf,
                                                   __bf16* __restrict__ outb,
                                                   int N, int K) {
    __shared__ alignas(64) __bf16 a_lds[64 * 32];    // [m][k]
    __shared__ alignas(64) __bf16 wt_lds[128 * 32];  // [n][k]
    const int tid = threadIdx.x, lane = tid & 31, wid = tid >> 5;
    const int wrow = wid >> 2, wcol = wid & 3;
    const int m0 = blockIdx.y * 64, n0 = blockIdx.x * 128;

    const int am = tid >> 2, ak = (tid & 3) * 8;       // A-tile slot (1x b128)
    const int wn = tid >> 1, wk = (tid & 1) * 16;      // W-tile slot (2x b128)
    const __bf16* aptr = A  + (size_t)(m0 + am) * K + ak;
    const __bf16* wptr = WT + (size_t)(n0 + wn) * K + wk;

    const v8f vzero = {0.f, 0.f, 0.f, 0.f, 0.f, 0.f, 0.f, 0.f};
    v8f acc[2][2];
#pragma unroll
    for (int mi = 0; mi < 2; ++mi)
#pragma unroll
        for (int ni = 0; ni < 2; ++ni) acc[mi][ni] = vzero;

    for (int k0 = 0; k0 < K; k0 += 32) {
        g2l_b128(a_lds + am * 32 + ak, aptr + k0);
        g2l_b128(wt_lds + wn * 32 + wk,     wptr + k0);
        g2l_b128(wt_lds + wn * 32 + wk + 8, wptr + k0 + 8);
        if (k0 + 32 < K) {   // prefetch next tiles into GL2 (global_prefetch_b8)
            __builtin_prefetch(aptr + k0 + 32, 0, 3);
            __builtin_prefetch(wptr + k0 + 32, 0, 3);
        }
        wait_lds_copies();
        __syncthreads();
        v16bf af[2], bfr[2];
        af[0]  = frag_a(a_lds, wrow * 32 + 0, 32, 0, lane);
        af[1]  = frag_a(a_lds, wrow * 32 + 16, 32, 0, lane);
        bfr[0] = frag_b(wt_lds, wcol * 32 + 0, 32, 0, lane);
        bfr[1] = frag_b(wt_lds, wcol * 32 + 16, 32, 0, lane);
#pragma unroll
        for (int mi = 0; mi < 2; ++mi)
#pragma unroll
            for (int ni = 0; ni < 2; ++ni)
                acc[mi][ni] = wmma_bf16(af[mi], bfr[ni], acc[mi][ni]);
        __syncthreads();
    }

    // epilogue: C/D layout: VGPR i -> row i (lanes 0-15) / i+8 (lanes 16-31),
    // N = lane&15 within each 16x16 tile.
    const int rbase = (lane & 16) ? 8 : 0, cil = lane & 15;
#pragma unroll
    for (int mi = 0; mi < 2; ++mi)
#pragma unroll
        for (int ni = 0; ni < 2; ++ni)
#pragma unroll
            for (int i = 0; i < 8; ++i) {
                int grow = m0 + wrow * 32 + mi * 16 + i + rbase;
                int gcol = n0 + wcol * 32 + ni * 16 + cil;
                size_t idx = (size_t)grow * N + gcol;
                float val = acc[mi][ni][i] + bias[gcol];
                if (MODE == 0) {
                    outb[idx] = (__bf16)val;
                } else if (MODE == 1) {
                    outf[idx] = aux[idx] + val;
                } else if (MODE == 2) {
                    float g = 0.5f * val * (1.0f + erff(val * 0.70710678118f));
                    outb[idx] = (__bf16)g;
                } else {   // MODE 3: V in [b, h, d, tok] layout for flash B-frags
                    int bb = grow >> 10, tok = grow & (NQ - 1);
                    int hh = gcol >> 6,  dd  = gcol & (HD - 1);
                    size_t ti = (((size_t)bb * HEADS + hh) * HD + dd) * NC + tok;
                    outb[ti] = (__bf16)val;
                }
            }
}

// ---------------------------------------------------------------------------
// Flash attention: grid (Nq/64, H, B), block 256 (8 waves, 4x2 wave grid).
// Per block: 64 q-rows x full Nc, online softmax, 64-key tiles.
// S = (Q K^T) * 0.125 + beta[h] * wind_bias  (wind read exactly once).
// V arrives pre-transposed as Vt[b][h][d][tok] -> straight b128 LDS fills.
// ---------------------------------------------------------------------------
__global__ void __launch_bounds__(256) k_flash_attn(const __bf16* __restrict__ Q,
                                                    const __bf16* __restrict__ Kb,
                                                    const __bf16* __restrict__ Vt,
                                                    const float* __restrict__ wind,
                                                    const float* __restrict__ beta,
                                                    __bf16* __restrict__ Ob) {
    __shared__ alignas(64) __bf16 q_lds[64 * 64];   // [q][d]
    __shared__ alignas(64) __bf16 k_lds[64 * 64];   // [key][d]  (B-frag, QK^T)
    __shared__ alignas(64) __bf16 vt_lds[64 * 64];  // [d][key]  (B-frag, P@V)
    __shared__ alignas(64) __bf16 p_lds[64 * 64];   // [q][key]  (A-frag, P@V)
    __shared__ float pmax[128], psum[128];          // [wcol][row]
    __shared__ float m_state[64], l_state[64], sc[64];

    const int tid = threadIdx.x, lane = tid & 31, wid = tid >> 5;
    const int wrow = wid >> 1, wcol = wid & 1;
    const int qt = blockIdx.x, h = blockIdx.y, b = blockIdx.z;
    const int rbase = (lane & 16) ? 8 : 0, cil = lane & 15;
    const float scale = 0.125f;   // hd^-0.5, hd = 64
    const float betah = beta[h];

    const int row = tid >> 2, c0 = (tid & 3) * 16;  // tile-fill slot (2x b128)
    {   // load Q tile once
        const __bf16* src = Q + ((size_t)b * NQ + qt * 64 + row) * DIM + h * HD + c0;
        g2l_b128(q_lds + row * 64 + c0,     src);
        g2l_b128(q_lds + row * 64 + c0 + 8, src + 8);
    }
    if (tid < 64) { m_state[tid] = -1e30f; l_state[tid] = 0.0f; }

    const __bf16* kbase = Kb + ((size_t)b * NC + row) * DIM + h * HD + c0;
    const __bf16* vbase = Vt + (((size_t)b * HEADS + h) * HD + row) * NC + c0;
    const int qrow0 = qt * 64 + wrow * 16 + rbase;
    const float* wbase = wind + ((size_t)b * NQ + qrow0) * NC + wcol * 32 + cil;

    const v8f vzero = {0.f, 0.f, 0.f, 0.f, 0.f, 0.f, 0.f, 0.f};
    v8f o[2]; o[0] = vzero; o[1] = vzero;

    for (int kt = 0; kt < NC / 64; ++kt) {
        __syncthreads();   // prior-iter LDS consumers done
        {   // K tile [key][d]; V tile [d][key] — both straight b128 copies
            const __bf16* ks = kbase + (size_t)(kt * 64) * DIM;
            g2l_b128(k_lds + row * 64 + c0,     ks);
            g2l_b128(k_lds + row * 64 + c0 + 8, ks + 8);
            const __bf16* vs = vbase + kt * 64;
            g2l_b128(vt_lds + row * 64 + c0,     vs);
            g2l_b128(vt_lds + row * 64 + c0 + 8, vs + 8);
        }
        if (kt + 1 < NC / 64)   // prefetch next wind_bias tile (largest stream)
            __builtin_prefetch(wbase + (kt + 1) * 64, 0, 2);
        wait_lds_copies();
        __syncthreads();

        // S tile: wave's 16 rows x 32 cols; K-dim = hd = 64 (2 wmma steps)
        v8f s[2]; s[0] = vzero; s[1] = vzero;
#pragma unroll
        for (int ks = 0; ks < 2; ++ks) {
            v16bf af = frag_a(q_lds, wrow * 16, 64, ks * 32, lane);
#pragma unroll
            for (int nt = 0; nt < 2; ++nt) {
                v16bf bfr = frag_b(k_lds, wcol * 32 + nt * 16, 64, ks * 32, lane);
                s[nt] = wmma_bf16(af, bfr, s[nt]);
            }
        }

        // scale + fused wind bias
        float sv[2][8];
#pragma unroll
        for (int nt = 0; nt < 2; ++nt)
#pragma unroll
            for (int i = 0; i < 8; ++i) {
                float wv = wbase[(size_t)i * NC + kt * 64 + nt * 16];
                sv[nt][i] = s[nt][i] * scale + betah * wv;
            }

        // row max over this wave's 32 cols (shuffle across 16 lanes of a half)
        float rmax[8];
#pragma unroll
        for (int i = 0; i < 8; ++i) rmax[i] = fmaxf(sv[0][i], sv[1][i]);
#pragma unroll
        for (int msk = 1; msk < 16; msk <<= 1)
#pragma unroll
            for (int i = 0; i < 8; ++i)
                rmax[i] = fmaxf(rmax[i], __shfl_xor(rmax[i], msk, 32));
        if (cil == 0)
#pragma unroll
            for (int i = 0; i < 8; ++i)
                pmax[wcol * 64 + wrow * 16 + rbase + i] = rmax[i];
        __syncthreads();

        if (tid < 64) {   // combine partials, update running max, rescale factor
            float mo = m_state[tid];
            float mn = fmaxf(mo, fmaxf(pmax[tid], pmax[64 + tid]));
            m_state[tid] = mn;
            sc[tid] = __expf(mo - mn);
        }
        __syncthreads();

        // P = exp(S - m) -> bf16 LDS; partial row sums
        float rsum[8];
#pragma unroll
        for (int i = 0; i < 8; ++i) rsum[i] = 0.0f;
#pragma unroll
        for (int nt = 0; nt < 2; ++nt)
#pragma unroll
            for (int i = 0; i < 8; ++i) {
                int r = wrow * 16 + rbase + i;
                float p = __expf(sv[nt][i] - m_state[r]);
                rsum[i] += p;
                p_lds[r * 64 + wcol * 32 + nt * 16 + cil] = (__bf16)p;
            }
#pragma unroll
        for (int msk = 1; msk < 16; msk <<= 1)
#pragma unroll
            for (int i = 0; i < 8; ++i)
                rsum[i] += __shfl_xor(rsum[i], msk, 32);
        if (cil == 0)
#pragma unroll
            for (int i = 0; i < 8; ++i)
                psum[wcol * 64 + wrow * 16 + rbase + i] = rsum[i];

        // rescale O accumulators for the new max
#pragma unroll
        for (int i = 0; i < 8; ++i) {
            float f = sc[wrow * 16 + rbase + i];
            o[0][i] *= f; o[1][i] *= f;
        }
        __syncthreads();   // p_lds + psum visible

        if (tid < 64)
            l_state[tid] = l_state[tid] * sc[tid] + psum[tid] + psum[64 + tid];

        // O += P @ V   (K-dim = 64 keys, 2 wmma steps)
#pragma unroll
        for (int ks = 0; ks < 2; ++ks) {
            v16bf af = frag_a(p_lds, wrow * 16, 64, ks * 32, lane);
#pragma unroll
            for (int nt = 0; nt < 2; ++nt) {
                v16bf bfr = frag_b(vt_lds, wcol * 32 + nt * 16, 64, ks * 32, lane);
                o[nt] = wmma_bf16(af, bfr, o[nt]);
            }
        }
    }

    __syncthreads();
    if (tid < 64) sc[tid] = 1.0f / l_state[tid];
    __syncthreads();
#pragma unroll
    for (int nt = 0; nt < 2; ++nt)
#pragma unroll
        for (int i = 0; i < 8; ++i) {
            int r = wrow * 16 + rbase + i;
            float val = o[nt][i] * sc[r];
            Ob[((size_t)b * NQ + qt * 64 + r) * DIM + h * HD + wcol * 32 + nt * 16 + cil] =
                (__bf16)val;
        }
}

// ---------------------------------------------------------------------------
// Host-side orchestration
// ---------------------------------------------------------------------------
extern "C" void kernel_launch(void* const* d_in, const int* in_sizes, int n_in,
                              void* d_out, int out_size, void* d_ws, size_t ws_size,
                              hipStream_t stream) {
    (void)in_sizes; (void)n_in; (void)out_size; (void)ws_size;
    const float* query   = (const float*)d_in[0];
    const float* context = (const float*)d_in[1];
    const float* wind    = (const float*)d_in[2];
    const float* ln_q_w  = (const float*)d_in[3];
    const float* ln_q_b  = (const float*)d_in[4];
    const float* ln_kv_w = (const float*)d_in[5];
    const float* ln_kv_b = (const float*)d_in[6];
    const float* Wq = (const float*)d_in[7];  const float* bq = (const float*)d_in[8];
    const float* Wk = (const float*)d_in[9];  const float* bk = (const float*)d_in[10];
    const float* Wv = (const float*)d_in[11]; const float* bv = (const float*)d_in[12];
    const float* Wo = (const float*)d_in[13]; const float* bo = (const float*)d_in[14];
    const float* beta = (const float*)d_in[15];
    const float* ln_f_w = (const float*)d_in[16];
    const float* ln_f_b = (const float*)d_in[17];
    const float* W1 = (const float*)d_in[18]; const float* b1 = (const float*)d_in[19];
    const float* W2 = (const float*)d_in[20]; const float* b2 = (const float*)d_in[21];
    float* out = (float*)d_out;

    // workspace carve-up (256B aligned)
    char* ws = (char*)d_ws;
    auto carve = [&ws](size_t bytes) -> void* {
        void* p = (void*)ws;
        ws += (bytes + 255) & ~(size_t)255;
        return p;
    };
    const size_t SZ_W   = (size_t)DIM * DIM;
    const size_t SZ_W1  = (size_t)DIM * HIDDEN;
    const size_t SZ_TOK = (size_t)MTOT * DIM;
    const size_t SZ_H1  = (size_t)MTOT * HIDDEN;

    __bf16* wqT = (__bf16*)carve(SZ_W * 2);    // [N][K] transposed bf16 weights
    __bf16* wkT = (__bf16*)carve(SZ_W * 2);
    __bf16* wvT = (__bf16*)carve(SZ_W * 2);
    __bf16* woT = (__bf16*)carve(SZ_W * 2);
    __bf16* w1T = (__bf16*)carve(SZ_W1 * 2);
    __bf16* w2T = (__bf16*)carve(SZ_W1 * 2);
    __bf16* lnq_b  = (__bf16*)carve(SZ_TOK * 2);
    __bf16* lnkv_b = (__bf16*)carve(SZ_TOK * 2);
    __bf16* q_b    = (__bf16*)carve(SZ_TOK * 2);
    __bf16* k_b    = (__bf16*)carve(SZ_TOK * 2);
    __bf16* vt_g   = (__bf16*)carve(SZ_TOK * 2);   // V in [b,h,d,tok]
    __bf16* attn_b = (__bf16*)carve(SZ_TOK * 2);
    __bf16* lnf_b  = (__bf16*)carve(SZ_TOK * 2);
    __bf16* h1_b   = (__bf16*)carve(SZ_H1 * 2);
    float*  x_f    = (float*)carve(SZ_TOK * 4);

    // 1) weight transpose + bf16 conversion: WT[N][K]
    dim3 tb(32, 8);
    k_transpose_bf16<<<dim3(DIM / 32, DIM / 32), tb, 0, stream>>>(Wq, wqT, DIM, DIM);
    k_transpose_bf16<<<dim3(DIM / 32, DIM / 32), tb, 0, stream>>>(Wk, wkT, DIM, DIM);
    k_transpose_bf16<<<dim3(DIM / 32, DIM / 32), tb, 0, stream>>>(Wv, wvT, DIM, DIM);
    k_transpose_bf16<<<dim3(DIM / 32, DIM / 32), tb, 0, stream>>>(Wo, woT, DIM, DIM);
    k_transpose_bf16<<<dim3(HIDDEN / 32, DIM / 32), tb, 0, stream>>>(W1, w1T, DIM, HIDDEN);
    k_transpose_bf16<<<dim3(DIM / 32, HIDDEN / 32), tb, 0, stream>>>(W2, w2T, HIDDEN, DIM);

    // 2) layernorms of inputs
    k_layernorm_bf16<<<MTOT, 256, 0, stream>>>(query,   ln_q_w,  ln_q_b,  lnq_b);
    k_layernorm_bf16<<<MTOT, 256, 0, stream>>>(context, ln_kv_w, ln_kv_b, lnkv_b);

    // 3) Q, K projections (bf16 out); V projection scattered to [b,h,d,tok]
    dim3 gP(DIM / 128, MTOT / 64);
    k_gemm_bf16<0><<<gP, 256, 0, stream>>>(lnq_b,  wqT, bq, nullptr, nullptr, q_b, DIM, DIM);
    k_gemm_bf16<0><<<gP, 256, 0, stream>>>(lnkv_b, wkT, bk, nullptr, nullptr, k_b, DIM, DIM);
    k_gemm_bf16<3><<<gP, 256, 0, stream>>>(lnkv_b, wvT, bv, nullptr, nullptr, vt_g, DIM, DIM);

    // 4) flash attention with fused beta*wind_bias
    dim3 gA(NQ / 64, HEADS, BATCH);
    k_flash_attn<<<gA, 256, 0, stream>>>(q_b, k_b, vt_g, wind, beta, attn_b);

    // 5) output projection + residual: x = query + attn@Wo + bo  (f32)
    k_gemm_bf16<1><<<gP, 256, 0, stream>>>(attn_b, woT, bo, query, x_f, nullptr, DIM, DIM);

    // 6) final LN, MLP up (GELU), MLP down + residual -> d_out
    k_layernorm_bf16<<<MTOT, 256, 0, stream>>>(x_f, ln_f_w, ln_f_b, lnf_b);
    dim3 g1(HIDDEN / 128, MTOT / 64);
    k_gemm_bf16<2><<<g1, 256, 0, stream>>>(lnf_b, w1T, b1, nullptr, nullptr, h1_b, HIDDEN, DIM);
    k_gemm_bf16<1><<<gP, 256, 0, stream>>>(h1_b, w2T, b2, x_f, out, nullptr, DIM, HIDDEN);
}
